// AttentionBlock_61014305407635
// MI455X (gfx1250) — matrix operand order repaired and checked by reading, not compiled, not run on hardware
//
#include <hip/hip_runtime.h>

typedef __attribute__((ext_vector_type(16))) _Float16 v16h;
typedef __attribute__((ext_vector_type(8)))  float    v8f;

typedef __attribute__((ext_vector_type(4))) unsigned int tdm_u32x4;
typedef __attribute__((ext_vector_type(8))) int          tdm_i32x8;
typedef __attribute__((ext_vector_type(4))) int          tdm_i32x4;

#define NPOS 1024
#define CCH  256
// LDS A-tile row stride in halfs: 256 data + 8 pad (TDM pad: 4 DWORDs / 512B)
#define APAD 264

union FragU { uint4 u[2]; v16h h; };

static __device__ __forceinline__ v16h ld_frag2(const void* p0, const void* p1) {
  FragU f;
  f.u[0] = *reinterpret_cast<const uint4*>(p0);
  f.u[1] = *reinterpret_cast<const uint4*>(p1);
  return f.h;
}

static __device__ __forceinline__ v8f wmma_f16(v16h a, v16h b, v8f c) {
  // 8 args: (neg_a, A, neg_b, B, c_mod, C, reuse_a, reuse_b)
  return __builtin_amdgcn_wmma_f32_16x16x32_f16(false, a, false, b, (short)0, c, false, false);
}

// ---------------------------------------------------------------------------
// TDM: DMA a 2D tile (rows x dim0_units, 8-byte units) from global into LDS,
// inserting 4 DWORDs of padding after every 128 DWORDs (512B row) so row-
// parallel b128 fragment reads from LDS spread across banks.
// Descriptor packing per CDNA5 ISA ch.8 (D# group0/group1).
// ---------------------------------------------------------------------------
#if __has_builtin(__builtin_amdgcn_tensor_load_to_lds)
#define HAVE_TDM 1
#endif

#ifdef HAVE_TDM
static __device__ __forceinline__ void tdm_load_2d_padded(
    unsigned int lds_off, const void* gptr, unsigned int dim0_units,
    unsigned int rows, unsigned int stride_units) {
  unsigned long long ga = (unsigned long long)(uintptr_t)gptr;
  tdm_u32x4 g0;
  g0.x = 1u;                                    // count=1 (valid user D#)
  g0.y = lds_off;                               // lds_addr (bytes)
  g0.z = (unsigned int)ga;                      // global_addr[31:0]
  g0.w = (unsigned int)((ga >> 32) & 0x01FFFFFFu) | 0x80000000u; // addr[56:32], type=2
  tdm_i32x8 g1;
  // data_size=3 (8B), pad_enable=1, pad_interval=6 (128 DW), pad_amount=3 (4 DW)
  g1[0] = (int)((3u << 16) | (1u << 20) | (6u << 22) | (3u << 25));
  g1[1] = (int)(dim0_units << 16);                       // tensor_dim0[15:0]
  g1[2] = (int)((dim0_units >> 16) | (rows << 16));      // dim0 hi, tensor_dim1 lo
  g1[3] = (int)((rows >> 16) | (dim0_units << 16));      // dim1 hi, tile_dim0
  g1[4] = (int)rows;                                     // tile_dim1 (tile_dim2=0 -> 2D)
  g1[5] = (int)stride_units;                             // tensor_dim0_stride[31:0]
  g1[6] = 0;                                             // stride hi, dim1_stride lo
  g1[7] = 0;
  tdm_i32x4 z4 = {0, 0, 0, 0};
#if __has_include(<hip/amd_detail/amd_gfx1250_TDM.h>)
  tdm_i32x8 z8 = {0, 0, 0, 0, 0, 0, 0, 0};
  __builtin_amdgcn_tensor_load_to_lds(g0, g1, z4, z4, z8, 0);   // clang-23 form
#else
  __builtin_amdgcn_tensor_load_to_lds(g0, g1, z4, z4, 0);       // ROCm 7.2 form
#endif
}
#endif

// Stage 32 rows x 256 halfs of A (row stride CCH halfs in global) into LDS
// with APAD row stride.  Called by all 256 threads; TDM path issues from
// wave 0 only and waits on TENSORcnt before the workgroup barrier.
static __device__ __forceinline__ void stage_a_tile(
    _Float16* lds, const _Float16* gsrc, int tid) {
#ifdef HAVE_TDM
  if ((tid >> 5) == 0) {
    tdm_load_2d_padded((unsigned int)(uintptr_t)lds, gsrc,
                       /*dim0 8B units*/ 64, /*rows*/ 32, /*stride units*/ 64);
    __builtin_amdgcn_s_wait_tensorcnt(0);
  }
#else
  for (int i = tid; i < 32 * 32; i += 256) {       // 32 rows x 32 uint4/row
    int row = i >> 5, c = i & 31;
    *reinterpret_cast<uint4*>(lds + (size_t)row * APAD + c * 8) =
        *reinterpret_cast<const uint4*>(gsrc + (size_t)row * CCH + c * 8);
  }
#endif
  __syncthreads();
}

// ---------------------------------------------------------------------------
// Kernel 0: convert the four 256x256 weight matrices to f16 (rows = out ch,
// contiguous in K=in ch, i.e. exactly the B^T layout WMMA wants).
// ---------------------------------------------------------------------------
__global__ void __launch_bounds__(256) cvt_w_kernel(
    const float* __restrict__ wq, const float* __restrict__ wk,
    const float* __restrict__ wv, const float* __restrict__ wo,
    _Float16* __restrict__ wqh, _Float16* __restrict__ wkh,
    _Float16* __restrict__ wvh, _Float16* __restrict__ woh) {
  int i = blockIdx.x * 256 + threadIdx.x;   // 65536 elements each
  wqh[i] = (_Float16)wq[i];
  wkh[i] = (_Float16)wk[i];
  wvh[i] = (_Float16)wv[i];
  woh[i] = (_Float16)wo[i];
}

// ---------------------------------------------------------------------------
// Kernel 1: GroupNorm.  One block per (batch, group); group = 8 ch x 1024.
// Writes f16 activations transposed to [B, N, C] (16B vector stores).
// ---------------------------------------------------------------------------
__global__ void __launch_bounds__(256) gn_kernel(
    const float* __restrict__ x, const float* __restrict__ gw,
    const float* __restrict__ gb, _Float16* __restrict__ hT) {
  const int b = blockIdx.y, g = blockIdx.x, tid = threadIdx.x;
  const float* base = x + (((size_t)b * CCH + g * 8) * NPOS);

  float s = 0.f, s2 = 0.f;
  for (int i = tid; i < 8 * NPOS; i += 256) {
    float v = base[i];
    s += v; s2 += v * v;
  }
  for (int m = 16; m >= 1; m >>= 1) {
    s  += __shfl_xor(s,  m);
    s2 += __shfl_xor(s2, m);
  }
  __shared__ float ws1[8], ws2[8];
  const int wid = tid >> 5, lane = tid & 31;
  if (lane == 0) { ws1[wid] = s; ws2[wid] = s2; }
  __syncthreads();
  s = 0.f; s2 = 0.f;
  for (int w = 0; w < 8; ++w) { s += ws1[w]; s2 += ws2[w]; }

  const float inv = 1.0f / (float)(8 * NPOS);
  const float mean = s * inv;
  const float var  = s2 * inv - mean * mean;
  const float rstd = rsqrtf(var + 1e-5f);

  float wgt[8], bia[8];
  #pragma unroll
  for (int cl = 0; cl < 8; ++cl) {
    int c = g * 8 + cl;
    wgt[cl] = gw[c] * rstd;
    bia[cl] = gb[c] - mean * wgt[cl];
  }
  #pragma unroll
  for (int j = 0; j < 4; ++j) {
    int npos = tid + 256 * j;
    _Float16 tmp[8];
    #pragma unroll
    for (int cl = 0; cl < 8; ++cl)
      tmp[cl] = (_Float16)(base[cl * NPOS + npos] * wgt[cl] + bia[cl]);
    *reinterpret_cast<uint4*>(hT + (((size_t)b * NPOS + npos) * CCH + g * 8)) =
        *reinterpret_cast<const uint4*>(tmp);
  }
}

// ---------------------------------------------------------------------------
// Kernel 2: WMMA GEMM  out[b] = A[b] (NxC) * BT^T (CxC) + bias
// Block covers rows [32*bx, 32*bx+32); A-tile staged into LDS via TDM and
// shared by the 4 waves per 16-row slice.  Each wave: 16(M) x 64(N) tile.
// trans==0: store [B,N,C] f16 (q,k).  trans==1: store [B,C,N] f16 (v^T).
// ---------------------------------------------------------------------------
__global__ void __launch_bounds__(256) gemm_nc_kernel(
    const _Float16* __restrict__ Abase, const _Float16* __restrict__ BT,
    const float* __restrict__ bias, _Float16* __restrict__ out, int trans) {
  const int b = blockIdx.y;
  const int tid = threadIdx.x, wid = tid >> 5, lane = tid & 31;
  const int ln = lane & 15, lh = lane >> 4;
  const int mt_local = wid >> 2, ng = wid & 3;
  const int m0 = blockIdx.x * 32 + mt_local * 16;
  const int nbase = ng * 64;

  __shared__ __align__(16) _Float16 Atile[32 * APAD];

  const _Float16* A = Abase + (size_t)b * NPOS * CCH + (size_t)blockIdx.x * 32 * CCH;
  stage_a_tile(Atile, A, tid);

  const _Float16* arow = Atile + (size_t)(mt_local * 16 + ln) * APAD;

  v8f acc[4] = {};
  #pragma unroll
  for (int kk = 0; kk < 8; ++kk) {
    const int k0 = kk * 32;
    v16h a = ld_frag2(arow + k0 + 16 * lh, arow + k0 + 16 * lh + 8);
    #pragma unroll
    for (int t = 0; t < 4; ++t) {
      const _Float16* brow = BT + (size_t)(nbase + t * 16 + ln) * CCH + k0 + 16 * lh;
      v16h bf = ld_frag2(brow, brow + 8);
      acc[t] = wmma_f16(a, bf, acc[t]);
    }
  }
  #pragma unroll
  for (int t = 0; t < 4; ++t) {
    const int col = nbase + t * 16 + ln;
    const float bv = bias[col];
    if (!trans) {
      _Float16* op = out + ((size_t)b * NPOS + m0 + 8 * lh) * CCH + col;
      #pragma unroll
      for (int r = 0; r < 8; ++r) op[(size_t)r * CCH] = (_Float16)(acc[t][r] + bv);
    } else {
      _Float16 tmp[8];
      #pragma unroll
      for (int r = 0; r < 8; ++r) tmp[r] = (_Float16)(acc[t][r] + bv);
      *reinterpret_cast<uint4*>(out + ((size_t)b * CCH + col) * NPOS + m0 + 8 * lh) =
          *reinterpret_cast<const uint4*>(tmp);
    }
  }
}

// ---------------------------------------------------------------------------
// Kernel 3: flash attention.  Block = (batch, 16-query tile), 8 waves.
// Wave w: QK^T for keys [16w,16w+16) of each 128-key block, P*V for output
// channels [32w, 32w+32).  Online softmax state + P tile shared via LDS.
// ---------------------------------------------------------------------------
__global__ void __launch_bounds__(256) attn_kernel(
    const _Float16* __restrict__ q, const _Float16* __restrict__ k,
    const _Float16* __restrict__ vT, _Float16* __restrict__ hout) {
  const int b = blockIdx.y;
  const int q0 = blockIdx.x * 16;
  const int tid = threadIdx.x, wid = tid >> 5, lane = tid & 31;
  const int ln = lane & 15, lh = lane >> 4;

  __shared__ __align__(16) _Float16 Ps[16][136];   // P tile, padded rows
  __shared__ float row_m[16], row_l[16], alpha_s[16], mnew_s[16];
  __shared__ float redmax[8][16], redsum[8][16];

  const _Float16* Qb = q  + ((size_t)b * NPOS + q0) * CCH;
  const _Float16* Kb = k  + (size_t)b * NPOS * CCH;
  const _Float16* Vb = vT + (size_t)b * CCH * NPOS;

  // Q fragments for all 8 K-steps stay in registers (64 VGPRs).
  v16h qf[8];
  {
    const _Float16* qr = Qb + (size_t)ln * CCH;
    #pragma unroll
    for (int kk = 0; kk < 8; ++kk)
      qf[kk] = ld_frag2(qr + kk * 32 + 16 * lh, qr + kk * 32 + 16 * lh + 8);
  }
  v8f o0 = {}, o1 = {};
  if (tid < 16) { row_m[tid] = -1e30f; row_l[tid] = 0.f; }
  __syncthreads();

  for (int blk = 0; blk < 8; ++blk) {
    const int m0 = blk * 128;
    // ---- S = (Q K^T) * scale, this wave's 16 keys ----
    v8f s = {};
    const _Float16* kr = Kb + (size_t)(m0 + wid * 16 + ln) * CCH + 16 * lh;
    #pragma unroll
    for (int kk = 0; kk < 8; ++kk) {
      v16h bf = ld_frag2(kr + kk * 32, kr + kk * 32 + 8);
      s = wmma_f16(qf[kk], bf, s);
    }
    float m8[8];
    #pragma unroll
    for (int r = 0; r < 8; ++r) { s[r] *= 0.0625f; m8[r] = s[r]; }
    // row max across the 16 key-lanes (stays inside each 16-lane half)
    #pragma unroll
    for (int msk = 8; msk >= 1; msk >>= 1)
      #pragma unroll
      for (int r = 0; r < 8; ++r) m8[r] = fmaxf(m8[r], __shfl_xor(m8[r], msk));
    if (ln == 0)
      #pragma unroll
      for (int r = 0; r < 8; ++r) redmax[wid][r + 8 * lh] = m8[r];
    __syncthreads();

    if (tid < 16) {
      float g = redmax[0][tid];
      #pragma unroll
      for (int w = 1; w < 8; ++w) g = fmaxf(g, redmax[w][tid]);
      const float mo = row_m[tid];
      const float mn = fmaxf(mo, g);
      mnew_s[tid]  = mn;
      alpha_s[tid] = __expf(mo - mn);
      row_m[tid]   = mn;
    }
    __syncthreads();

    // ---- P = exp(S - m_new); stage P into LDS; partial row sums ----
    float p8[8];
    #pragma unroll
    for (int r = 0; r < 8; ++r) {
      float pv = __expf(s[r] - mnew_s[r + 8 * lh]);
      p8[r] = pv;
      Ps[r + 8 * lh][wid * 16 + ln] = (_Float16)pv;
    }
    #pragma unroll
    for (int msk = 8; msk >= 1; msk >>= 1)
      #pragma unroll
      for (int r = 0; r < 8; ++r) p8[r] += __shfl_xor(p8[r], msk);
    if (ln == 0)
      #pragma unroll
      for (int r = 0; r < 8; ++r) redsum[wid][r + 8 * lh] = p8[r];
    // rescale running output accumulators
    #pragma unroll
    for (int r = 0; r < 8; ++r) {
      const float a = alpha_s[r + 8 * lh];
      o0[r] *= a; o1[r] *= a;
    }
    __syncthreads();

    if (tid < 16) {
      float acc = 0.f;
      #pragma unroll
      for (int w = 0; w < 8; ++w) acc += redsum[w][tid];
      row_l[tid] = alpha_s[tid] * row_l[tid] + acc;
    }

    // ---- O += P * V  (this wave's 32 output channels) ----
    const int c0 = wid * 32;
    const _Float16* v0r = Vb + (size_t)(c0 + ln) * NPOS + m0 + 16 * lh;
    const _Float16* v1r = Vb + (size_t)(c0 + 16 + ln) * NPOS + m0 + 16 * lh;
    #pragma unroll
    for (int kk = 0; kk < 4; ++kk) {
      v16h pa = ld_frag2(&Ps[ln][kk * 32 + 16 * lh], &Ps[ln][kk * 32 + 16 * lh + 8]);
      v16h b0 = ld_frag2(v0r + kk * 32, v0r + kk * 32 + 8);
      o0 = wmma_f16(pa, b0, o0);
      v16h b1 = ld_frag2(v1r + kk * 32, v1r + kk * 32 + 8);
      o1 = wmma_f16(pa, b1, o1);
    }
    __syncthreads();
  }

  // ---- normalize and store hout [B, N, C] f16 ----
  #pragma unroll
  for (int r = 0; r < 8; ++r) {
    const int row = r + 8 * lh;
    const float inv = 1.0f / row_l[row];
    _Float16* hp = hout + ((size_t)b * NPOS + q0 + row) * CCH;
    hp[wid * 32 + ln]      = (_Float16)(o0[r] * inv);
    hp[wid * 32 + 16 + ln] = (_Float16)(o1[r] * inv);
  }
}

// ---------------------------------------------------------------------------
// Kernel 4: out-projection + bias + residual, f32 output in [B, C, N].
// Same TDM-staged A-tile structure as the QKV GEMM.
// ---------------------------------------------------------------------------
__global__ void __launch_bounds__(256) outproj_kernel(
    const _Float16* __restrict__ Hbase, const _Float16* __restrict__ BT,
    const float* __restrict__ bias, const float* __restrict__ x,
    float* __restrict__ out) {
  const int b = blockIdx.y;
  const int tid = threadIdx.x, wid = tid >> 5, lane = tid & 31;
  const int ln = lane & 15, lh = lane >> 4;
  const int mt_local = wid >> 2, ng = wid & 3;
  const int m0 = blockIdx.x * 32 + mt_local * 16;
  const int nbase = ng * 64;

  __shared__ __align__(16) _Float16 Atile[32 * APAD];

  const _Float16* A = Hbase + (size_t)b * NPOS * CCH + (size_t)blockIdx.x * 32 * CCH;
  stage_a_tile(Atile, A, tid);

  const _Float16* arow = Atile + (size_t)(mt_local * 16 + ln) * APAD;

  v8f acc[4] = {};
  #pragma unroll
  for (int kk = 0; kk < 8; ++kk) {
    const int k0 = kk * 32;
    v16h a = ld_frag2(arow + k0 + 16 * lh, arow + k0 + 16 * lh + 8);
    #pragma unroll
    for (int t = 0; t < 4; ++t) {
      const _Float16* brow = BT + (size_t)(nbase + t * 16 + ln) * CCH + k0 + 16 * lh;
      v16h bf = ld_frag2(brow, brow + 8);
      acc[t] = wmma_f16(a, bf, acc[t]);
    }
  }
  #pragma unroll
  for (int t = 0; t < 4; ++t) {
    const int c = nbase + t * 16 + ln;
    const float bv = bias[c];
    const size_t off = ((size_t)b * CCH + c) * NPOS + m0 + 8 * lh;
    const float4* xp = reinterpret_cast<const float4*>(x + off);
    float4 x0 = xp[0], x1 = xp[1];
    float4 r0, r1;
    r0.x = x0.x + acc[t][0] + bv; r0.y = x0.y + acc[t][1] + bv;
    r0.z = x0.z + acc[t][2] + bv; r0.w = x0.w + acc[t][3] + bv;
    r1.x = x1.x + acc[t][4] + bv; r1.y = x1.y + acc[t][5] + bv;
    r1.z = x1.z + acc[t][6] + bv; r1.w = x1.w + acc[t][7] + bv;
    float4* op = reinterpret_cast<float4*>(out + off);
    op[0] = r0; op[1] = r1;
  }
}

// ---------------------------------------------------------------------------
extern "C" void kernel_launch(void* const* d_in, const int* in_sizes, int n_in,
                              void* d_out, int out_size, void* d_ws, size_t ws_size,
                              hipStream_t stream) {
  (void)in_sizes; (void)n_in; (void)out_size; (void)ws_size;
  const float* x   = (const float*)d_in[0];
  const float* gnw = (const float*)d_in[1];
  const float* gnb = (const float*)d_in[2];
  const float* wq  = (const float*)d_in[3];
  const float* bq  = (const float*)d_in[4];
  const float* wk  = (const float*)d_in[5];
  const float* bk  = (const float*)d_in[6];
  const float* wv  = (const float*)d_in[7];
  const float* bv  = (const float*)d_in[8];
  const float* wo  = (const float*)d_in[9];
  const float* bo  = (const float*)d_in[10];
  float* out = (float*)d_out;

  char* w = (char*)d_ws;
  const size_t SZ = (size_t)32 * NPOS * CCH * sizeof(_Float16);   // 16 MiB each
  _Float16* hT  = (_Float16*)(w);            // groupnormed act, [B,N,C]
  _Float16* qb  = (_Float16*)(w + SZ);       // q, [B,N,C]
  _Float16* kb  = (_Float16*)(w + 2 * SZ);   // k, [B,N,C]
  _Float16* vTb = (_Float16*)(w + 3 * SZ);   // v^T, [B,C,N]
  _Float16* ho  = (_Float16*)(w + 4 * SZ);   // attention out, [B,N,C]
  _Float16* wqh = (_Float16*)(w + 5 * SZ);   // f16 weights, 128 KiB each
  _Float16* wkh = wqh + 65536;
  _Float16* wvh = wkh + 65536;
  _Float16* woh = wvh + 65536;

  cvt_w_kernel<<<256, 256, 0, stream>>>(wq, wk, wv, wo, wqh, wkh, wvh, woh);
  gn_kernel<<<dim3(32, 32), 256, 0, stream>>>(x, gnw, gnb, hT);
  gemm_nc_kernel<<<dim3(32, 32), 256, 0, stream>>>(hT, wqh, bq, qb, 0);
  gemm_nc_kernel<<<dim3(32, 32), 256, 0, stream>>>(hT, wkh, bk, kb, 0);
  gemm_nc_kernel<<<dim3(32, 32), 256, 0, stream>>>(hT, wvh, bv, vTb, 1);
  attn_kernel<<<dim3(64, 32), 256, 0, stream>>>(qb, kb, vTb, ho);
  outproj_kernel<<<dim3(32, 32), 256, 0, stream>>>(ho, woh, bo, x, out);
}
